// GCN_9998683865367
// MI455X (gfx1250) — compile-verified, matrix operand order checked by reading, not run on hardware
//
#include <hip/hip_runtime.h>
#include <hip/hip_bf16.h>

#define N_NODES 50000
#define N_EDGES 1600000
#define IN_C 256
#define HID 128
#define OUT_C 64

typedef __attribute__((ext_vector_type(16))) __bf16 v16bf;
typedef __attribute__((ext_vector_type(8)))  float  v8f;

union BF16x16 { v16bf v; unsigned int d[8]; unsigned short u[16]; };

__device__ __forceinline__ unsigned short f32_to_bf16_bits(float f) {
    unsigned int x = __float_as_uint(f);
    x += 0x7FFFu + ((x >> 16) & 1u);   // round-to-nearest-even
    return (unsigned short)(x >> 16);
}
__device__ __forceinline__ unsigned int pack_bf16x2(float a, float b) {
    return (unsigned int)f32_to_bf16_bits(a) |
           ((unsigned int)f32_to_bf16_bits(b) << 16);
}

// ---------------------------------------------------------------------------
// Dense GEMM: C[MxN] = A[MxK](f32) * W[KxN](f32); bf16 WMMA, f32 accumulate.
// blockDim=256 = 8 wave32 waves; each wave owns one 16x16 output tile.
// K-step = 64 -> two v_wmma per barrier round. All tile params compile-time.
// ---------------------------------------------------------------------------
template <int K, int N, int ROWS>
__global__ __launch_bounds__(256) void gemm_bf16_wmma(
    const float* __restrict__ A, const float* __restrict__ W,
    float* __restrict__ C, int M)
{
    constexpr int NT = N / 16;              // n-tiles (must divide 8)
    static_assert(NT * (ROWS / 16) == 8, "8 waves must cover ROWS x N");
    constexpr int AS = 68;                  // LDS row stride in ushorts (136B, 8B-aligned)

    __shared__ unsigned short aT[ROWS * AS]; // A strip: ROWS x 64 (bf16)
    __shared__ unsigned short bT[N * AS];    // W tile transposed: N x 64 (bf16)

    const int tid  = threadIdx.x;
    const int lane = tid & 31;
    const int wave = tid >> 5;
    const int rowBase  = blockIdx.x * ROWS;
    const int mt       = wave / NT;
    const int nt       = wave % NT;
    const int laneHalf = lane >> 4;          // 0|1
    const int laneM    = lane & 15;          // M (A) / N (B,C) within tile
    const int hiK      = laneHalf << 3;      // lanes 16-31 -> K+8 (ISA 7.12.2)

    v8f acc = {0.f, 0.f, 0.f, 0.f, 0.f, 0.f, 0.f, 0.f};

    for (int kb = 0; kb < K; kb += 64) {
        // --- A strip: ROWS x 64 f32, float4 loads, packed-bf16 b64 stores ---
#pragma unroll
        for (int idx = tid; idx < ROWS * 16; idx += 256) {
            int r = idx >> 4, c4 = idx & 15;
            int gr = rowBase + r;
            if (gr >= M) gr = M - 1;         // clamp: output store is guarded
            const float4 t = *(const float4*)(A + (long)gr * K + kb + c4 * 4);
            uint2 p; p.x = pack_bf16x2(t.x, t.y); p.y = pack_bf16x2(t.z, t.w);
            *(uint2*)(&aT[r * AS + c4 * 4]) = p;
        }
        // --- W tile: 64 x N f32, float4 loads, transposed bf16 stores -------
        constexpr int NV = N / 4;
#pragma unroll
        for (int idx = tid; idx < 64 * NV; idx += 256) {
            int k = idx / NV, n4 = idx - k * NV;   // NV is power of two -> shifts
            const float4 t = *(const float4*)(W + (long)(kb + k) * N + n4 * 4);
            bT[(n4 * 4 + 0) * AS + k] = f32_to_bf16_bits(t.x);
            bT[(n4 * 4 + 1) * AS + k] = f32_to_bf16_bits(t.y);
            bT[(n4 * 4 + 2) * AS + k] = f32_to_bf16_bits(t.z);
            bT[(n4 * 4 + 3) * AS + k] = f32_to_bf16_bits(t.w);
        }
        __syncthreads();

        // --- fragments per ISA 16-bit layout; two 32-wide slabs -------------
        const unsigned int* ap = (const unsigned int*)&aT[(mt * 16 + laneM) * AS];
        const unsigned int* bp = (const unsigned int*)&bT[(nt * 16 + laneM) * AS];
        BF16x16 a0, a1, b0, b1;
#pragma unroll
        for (int i = 0; i < 8; ++i) {
            int pk = (((i & 4) << 2) + hiK + ((i & 3) << 1)) >> 1; // uint index
            a0.d[i] = ap[pk];      a1.d[i] = ap[pk + 16];
            b0.d[i] = bp[pk];      b1.d[i] = bp[pk + 16];
        }
        // EXEC uniform here; two WMMAs per round
        acc = __builtin_amdgcn_wmma_f32_16x16x32_bf16(
                  false, a0.v, false, b0.v, (short)0, acc, false, false);
        acc = __builtin_amdgcn_wmma_f32_16x16x32_bf16(
                  false, a1.v, false, b1.v, (short)0, acc, false, false);
        __syncthreads();
    }

    // C/D layout: lane -> N=lane&15; VGPR r -> M = r (+8 for lanes 16-31)
    const int gn = nt * 16 + laneM;
#pragma unroll
    for (int r = 0; r < 8; ++r) {
        int gm = rowBase + mt * 16 + r + (laneHalf << 3);
        if (gm < M) C[(long)gm * N + gn] = acc[r];
    }
}

// ---------------------------------------------------------------------------
// Degree / normalization
// ---------------------------------------------------------------------------
__global__ void init_deg(float* __restrict__ deg) {
    int i = blockIdx.x * blockDim.x + threadIdx.x;
    if (i < N_NODES) deg[i] = 1.0f;                 // self-loop
}

__global__ void deg_edges(const int* __restrict__ col, float* __restrict__ deg) {
    int e = blockIdx.x * blockDim.x + threadIdx.x;
    if (e < N_EDGES) atomicAdd(&deg[col[e]], 1.0f);
}

__global__ void deg_to_dinv(float* __restrict__ deg) {
    int i = blockIdx.x * blockDim.x + threadIdx.x;
    if (i < N_NODES) deg[i] = rsqrtf(deg[i]);       // deg >= 1 always
}

// agg[i,c] = h[i,c]*dinv[i]^2 + bias[c]  (self-loop + bias folded into init)
template <int C>
__global__ void init_agg(const float* __restrict__ h, const float* __restrict__ dinv,
                         const float* __restrict__ bias, float* __restrict__ agg)
{
    long i = (long)blockIdx.x * blockDim.x + threadIdx.x;
    if (i < (long)N_NODES * C) {
        int node = (int)(i / C);                    // C compile-time -> shift
        int ch   = (int)(i & (C - 1));
        float d  = dinv[node];
        agg[i] = h[i] * d * d + bias[ch];
    }
}

// one wave per edge; vectorized gather + f32 atomic scatter
template <int C>
__global__ __launch_bounds__(256) void edge_aggregate(
    const int* __restrict__ rowp, const int* __restrict__ colp,
    const float* __restrict__ h, const float* __restrict__ dinv,
    float* __restrict__ agg)
{
    long gtid = (long)blockIdx.x * blockDim.x + threadIdx.x;
    int e = (int)(gtid >> 5);
    if (e >= N_EDGES) return;
    int lane = threadIdx.x & 31;
    int r = rowp[e], c = colp[e];
    float nrm = dinv[r] * dinv[c];
    constexpr int V = C / 32;                        // floats per lane (4 or 2)
    const float* hs = h   + (long)r * C + lane * V;
    float*       ad = agg + (long)c * C + lane * V;
    if constexpr (V == 4) {
        float4 t = *(const float4*)hs;               // global_load_b128
        atomicAdd(ad + 0, t.x * nrm);
        atomicAdd(ad + 1, t.y * nrm);
        atomicAdd(ad + 2, t.z * nrm);
        atomicAdd(ad + 3, t.w * nrm);
    } else {
        float2 t = *(const float2*)hs;               // global_load_b64
        atomicAdd(ad + 0, t.x * nrm);
        atomicAdd(ad + 1, t.y * nrm);
    }
}

__global__ void relu_inplace(float* __restrict__ x, long n) {
    long i = (long)blockIdx.x * blockDim.x + threadIdx.x;
    if (i < n) x[i] = fmaxf(x[i], 0.0f);
}

// ---------------------------------------------------------------------------
extern "C" void kernel_launch(void* const* d_in, const int* in_sizes, int n_in,
                              void* d_out, int out_size, void* d_ws, size_t ws_size,
                              hipStream_t stream)
{
    const float* x  = (const float*)d_in[0];
    const int*   ei = (const int*)d_in[1];
    const float* W1 = (const float*)d_in[2];
    const float* b1 = (const float*)d_in[3];
    const float* W2 = (const float*)d_in[4];
    const float* b2 = (const float*)d_in[5];
    float* out = (float*)d_out;

    const int* rowp = ei;            // edge_index[0]  (source)
    const int* colp = ei + N_EDGES;  // edge_index[1]  (target)

    float* ws   = (float*)d_ws;
    float* dinv = ws;                         // N_NODES (deg -> dinv in place)
    float* h1   = ws + 50048;                 // N_NODES*HID
    float* agg1 = h1 + (long)N_NODES * HID;   // N_NODES*HID (-> relu(h))
    float* h2   = h1;                         // reuse h1 after layer-1 scatter

    // ---- normalization -----------------------------------------------------
    init_deg   <<<(N_NODES + 255) / 256, 256, 0, stream>>>(dinv);
    deg_edges  <<<(N_EDGES + 255) / 256, 256, 0, stream>>>(colp, dinv);
    deg_to_dinv<<<(N_NODES + 255) / 256, 256, 0, stream>>>(dinv);

    // ---- layer 1: h1 = x @ W1 ; agg1 = scatter(h1) + b1 ; relu -------------
    gemm_bf16_wmma<IN_C, HID, 16><<<(N_NODES + 15) / 16, 256, 0, stream>>>(
        x, W1, h1, N_NODES);
    init_agg<HID><<<(N_NODES * HID + 255) / 256, 256, 0, stream>>>(h1, dinv, b1, agg1);
    edge_aggregate<HID><<<(N_EDGES + 7) / 8, 256, 0, stream>>>(rowp, colp, h1, dinv, agg1);
    relu_inplace<<<(int)(((long)N_NODES * HID + 255) / 256), 256, 0, stream>>>(
        agg1, (long)N_NODES * HID);

    // ---- layer 2: h2 = relu @ W2 ; out = scatter(h2) + b2 ------------------
    gemm_bf16_wmma<HID, OUT_C, 32><<<(N_NODES + 31) / 32, 256, 0, stream>>>(
        agg1, W2, h2, N_NODES);
    init_agg<OUT_C><<<(N_NODES * OUT_C + 255) / 256, 256, 0, stream>>>(h2, dinv, b2, out);
    edge_aggregate<OUT_C><<<(N_EDGES + 7) / 8, 256, 0, stream>>>(rowp, colp, h2, dinv, out);
}
